// NatureCNN_10496900071444
// MI455X (gfx1250) — compile-verified, matrix operand order checked by reading, not run on hardware
//
#include <hip/hip_runtime.h>
#include <hip/hip_bf16.h>
#include <stdint.h>

// ---------------- CDNA5 WMMA helpers (gfx1250, wave32) ----------------
typedef __attribute__((ext_vector_type(16))) _Float16 v16h;
typedef __attribute__((ext_vector_type(8)))  _Float16 v8h;
typedef __attribute__((ext_vector_type(8)))  float    v8f;
typedef __attribute__((ext_vector_type(4)))  float    v4f;

#define DEV static __device__ __forceinline__

DEV v16h mk16(v8h lo, v8h hi) {
  return __builtin_shufflevector(lo, hi, 0,1,2,3,4,5,6,7,8,9,10,11,12,13,14,15);
}

// A/B fragment for v_wmma_f32_16x16x32_f16 from a row-major [rows x ld] f16 array.
// ISA 7.12.2: lanes 0-15 hold row m=lane, K chunks {kb..kb+7, kb+16..kb+23};
// lanes 16-31 hold row m=lane-16, K chunks {kb+8..kb+15, kb+24..kb+31}.
DEV v16h frag_rm(const _Float16* base, int ld, int row0, int kb, int lane) {
  int m  = lane & 15;
  int k1 = kb + ((lane & 16) ? 8 : 0);
  const _Float16* p = base + (size_t)(row0 + m) * (size_t)ld + k1;
  v8h lo = *(const v8h*)p;          // 16B aligned by layout construction
  v8h hi = *(const v8h*)(p + 16);
  return mk16(lo, hi);
}

DEV v8f wmma16(v16h a, v16h b, v8f c) {
  return __builtin_amdgcn_wmma_f32_16x16x32_f16(false, a, false, b, (short)0, c, false, false);
}

DEV float sigf(float x)      { return 1.f / (1.f + __expf(-x)); }
DEV float tanh_fast(float x) { float e = __expf(-2.f * x); return (1.f - e) / (1.f + e); }

// ---------------- problem constants ----------------
constexpr int NTOT   = 2048;   // 2*B*S
constexpr int DDIM   = 1500;
constexpr int DPAD   = 1504;   // 47*32
constexpr int FEATD  = 1536;   // 1024 conv + 512 mlp
constexpr int HH     = 100;
constexpr int HPAD   = 128;
constexpr int G4     = 400;    // 4*H
constexpr int GPAD   = 416;    // 4*H padded to 2x16-tile multiple (13 pairs)
constexpr int OUTN   = 130;
constexpr int OUTPAD = 144;
constexpr int FC1K   = 224;    // 200 padded

// ---------------- packing / conversion kernels ----------------
__global__ __launch_bounds__(256)
void k_cvt_pad(_Float16* __restrict__ dst, const float* __restrict__ src,
               int rowsD, int rowsS, int Kd, int Ks) {
  size_t tot = (size_t)rowsD * Kd;
  for (size_t i = (size_t)blockIdx.x * blockDim.x + threadIdx.x; i < tot;
       i += (size_t)gridDim.x * blockDim.x) {
    int r = (int)(i / Kd), k = (int)(i % Kd);
    float v = (r < rowsS && k < Ks) ? src[(size_t)r * Ks + k] : 0.f;
    dst[i] = (_Float16)v;
  }
}

// OIHW -> O x (HW*C): chunks of 8 K-values become channel-contiguous (NHWC acts)
__global__ __launch_bounds__(256)
void k_pack_hwio(_Float16* __restrict__ dst, const float* __restrict__ src,
                 int O, int C, int KH, int KW) {
  int tot = O * C * KH * KW;
  for (int i = blockIdx.x * blockDim.x + threadIdx.x; i < tot;
       i += gridDim.x * blockDim.x) {
    int o = i / (C * KH * KW);
    int rem = i % (C * KH * KW);
    int cell = rem / C, c = rem % C;
    int ky = cell / KW, kx = cell % KW;
    dst[i] = (_Float16)src[(((size_t)o * C + c) * KH + ky) * KW + kx];
  }
}

__global__ __launch_bounds__(256)
void k_bias_pad(float* __restrict__ d, const float* __restrict__ a,
                const float* __restrict__ b, int n, int npad) {
  int i = blockIdx.x * blockDim.x + threadIdx.x;
  if (i < npad) d[i] = (i < n) ? (a[i] + b[i]) : 0.f;
}

// ---------------- store helper ----------------
DEV void store_tile(void* Dst, int ldd, int row0, int ncol, int m0, v8f c, int mode) {
  if (mode == 2) {
    float* D = (float*)Dst;
#pragma unroll
    for (int j = 0; j < 8; j++)
      D[(size_t)(row0 + m0 + j) * ldd + ncol] = c[j];
  } else {
    _Float16* D = (_Float16*)Dst;
#pragma unroll
    for (int j = 0; j < 8; j++) {
      float v = c[j];
      if (mode == 1 && v < 0.f) v = 0.f;
      D[(size_t)(row0 + m0 + j) * ldd + ncol] = (_Float16)v;
    }
  }
}

// ---------------- generic f16 WMMA GEMM, 2x2 register blocking ----------------
// Each wave owns a 32x32 output block: fragments reused 2x -> 1 frag load / WMMA.
// D[M,N] = A[M,K] * Bt[N,K]^T (+bias); mode 0: f16 out, 1: f16+relu, 2: f32 out
__global__ __launch_bounds__(128)
void k_gemm2(const _Float16* __restrict__ A, int lda,
             const _Float16* __restrict__ Bt, int ldb,
             const float* __restrict__ bias,
             void* __restrict__ Dst, int ldd,
             int Mtiles2, int K, int mode) {
  int lane = threadIdx.x & 31;
  int mt = blockIdx.y * 4 + (threadIdx.x >> 5);
  if (mt >= Mtiles2) return;
  int row0 = mt * 32, n0 = blockIdx.x * 32;
  int ncol = lane & 15, m0 = (lane & 16) ? 8 : 0;
  float bv0 = bias ? bias[n0 + ncol] : 0.f;
  float bv1 = bias ? bias[n0 + 16 + ncol] : 0.f;
  v8f c00, c01, c10, c11;
#pragma unroll
  for (int j = 0; j < 8; j++) { c00[j] = bv0; c01[j] = bv1; c10[j] = bv0; c11[j] = bv1; }
  for (int kb = 0; kb < K; kb += 32) {
    __builtin_prefetch(A + (size_t)(row0 + ncol) * lda + kb + 128, 0, 1);
    v16h a0 = frag_rm(A, lda, row0,      kb, lane);
    v16h a1 = frag_rm(A, lda, row0 + 16, kb, lane);
    v16h b0 = frag_rm(Bt, ldb, n0,       kb, lane);
    v16h b1 = frag_rm(Bt, ldb, n0 + 16,  kb, lane);
    c00 = wmma16(a0, b0, c00);
    c01 = wmma16(a0, b1, c01);
    c10 = wmma16(a1, b0, c10);
    c11 = wmma16(a1, b1, c11);
  }
  store_tile(Dst, ldd, row0,      n0 + ncol,      m0, c00, mode);
  store_tile(Dst, ldd, row0,      n0 + 16 + ncol, m0, c01, mode);
  store_tile(Dst, ldd, row0 + 16, n0 + ncol,      m0, c10, mode);
  store_tile(Dst, ldd, row0 + 16, n0 + 16 + ncol, m0, c11, mode);
}

// ---------------- conv kernels (implicit im2col, WMMA, 2x2 blocked) ----------------
// conv1: obs NCHW f32 [n][3][64][64] -> a1 NHWC f16 [n][15][15][32], k8 s4, K=192
DEV v8h ld8_c1(const float* obs, int r, int kk) {
  int n = r / 225, s2 = r % 225;
  int oy = s2 / 15, ox = s2 % 15;
  int c = kk >> 6, ky = (kk >> 3) & 7;   // K = c*64 + ky*8 + kx (kx contiguous)
  const float* p = obs + ((((size_t)n * 3 + c) * 64 + oy * 4 + ky) * 64 + ox * 4);
  v4f f0 = *(const v4f*)p, f1 = *(const v4f*)(p + 4);
  v8h h;
#pragma unroll
  for (int i = 0; i < 4; i++) { h[i] = (_Float16)f0[i]; h[4 + i] = (_Float16)f1[i]; }
  return h;
}

__global__ __launch_bounds__(128)
void k_conv1(const float* __restrict__ obs, const _Float16* __restrict__ W,
             const float* __restrict__ bias, _Float16* __restrict__ a1) {
  int lane = threadIdx.x & 31;
  int mt = blockIdx.y * 4 + (threadIdx.x >> 5);
  int row0 = mt * 32, n0 = blockIdx.x * 32;   // n0 == 0 (N=32)
  int ncol = lane & 15, m0 = (lane & 16) ? 8 : 0;
  int r0 = row0 + (lane & 15), r1 = r0 + 16;
  float bv0 = bias[n0 + ncol], bv1 = bias[n0 + 16 + ncol];
  v8f c00, c01, c10, c11;
#pragma unroll
  for (int j = 0; j < 8; j++) { c00[j] = bv0; c01[j] = bv1; c10[j] = bv0; c11[j] = bv1; }
  for (int kb = 0; kb < 192; kb += 32) {
    int k1 = kb + ((lane & 16) ? 8 : 0);
    v16h a0 = mk16(ld8_c1(obs, r0, k1), ld8_c1(obs, r0, k1 + 16));
    v16h a1 = mk16(ld8_c1(obs, r1, k1), ld8_c1(obs, r1, k1 + 16));
    v16h b0 = frag_rm(W, 192, n0,      kb, lane);
    v16h b1 = frag_rm(W, 192, n0 + 16, kb, lane);
    c00 = wmma16(a0, b0, c00);
    c01 = wmma16(a0, b1, c01);
    c10 = wmma16(a1, b0, c10);
    c11 = wmma16(a1, b1, c11);
  }
  store_tile(a1, 32, row0,      n0 + ncol,      m0, c00, 1);
  store_tile(a1, 32, row0,      n0 + 16 + ncol, m0, c01, 1);
  store_tile(a1, 32, row0 + 16, n0 + ncol,      m0, c10, 1);
  store_tile(a1, 32, row0 + 16, n0 + 16 + ncol, m0, c11, 1);
}

// conv2: a1 NHWC [n][15][15][32] -> a2 NHWC [n][6][6][64], k4 s2, K=512 (HWIO)
DEV v8h ld8_c2(const _Float16* a1, int r, int kk) {
  int n = r / 36, s2 = r % 36;
  int oy = s2 / 6, ox = s2 % 6;
  int kcell = kk >> 5, c0 = kk & 31;
  int ky = kcell >> 2, kx = kcell & 3;
  return *(const v8h*)(a1 + (((size_t)(n * 15 + oy * 2 + ky) * 15) + (ox * 2 + kx)) * 32 + c0);
}

__global__ __launch_bounds__(128)
void k_conv2(const _Float16* __restrict__ a1, const _Float16* __restrict__ W,
             const float* __restrict__ bias, _Float16* __restrict__ a2) {
  int lane = threadIdx.x & 31;
  int mt = blockIdx.y * 4 + (threadIdx.x >> 5);
  int row0 = mt * 32, n0 = blockIdx.x * 32;
  int ncol = lane & 15, m0 = (lane & 16) ? 8 : 0;
  int r0 = row0 + (lane & 15), r1 = r0 + 16;
  float bv0 = bias[n0 + ncol], bv1 = bias[n0 + 16 + ncol];
  v8f c00, c01, c10, c11;
#pragma unroll
  for (int j = 0; j < 8; j++) { c00[j] = bv0; c01[j] = bv1; c10[j] = bv0; c11[j] = bv1; }
  for (int kb = 0; kb < 512; kb += 32) {
    int k1 = kb + ((lane & 16) ? 8 : 0);
    v16h a0 = mk16(ld8_c2(a1, r0, k1), ld8_c2(a1, r0, k1 + 16));
    v16h a1f = mk16(ld8_c2(a1, r1, k1), ld8_c2(a1, r1, k1 + 16));
    v16h b0 = frag_rm(W, 512, n0,      kb, lane);
    v16h b1 = frag_rm(W, 512, n0 + 16, kb, lane);
    c00 = wmma16(a0,  b0, c00);
    c01 = wmma16(a0,  b1, c01);
    c10 = wmma16(a1f, b0, c10);
    c11 = wmma16(a1f, b1, c11);
  }
  store_tile(a2, 64, row0,      n0 + ncol,      m0, c00, 1);
  store_tile(a2, 64, row0,      n0 + 16 + ncol, m0, c01, 1);
  store_tile(a2, 64, row0 + 16, n0 + ncol,      m0, c10, 1);
  store_tile(a2, 64, row0 + 16, n0 + 16 + ncol, m0, c11, 1);
}

// conv3: a2 NHWC [n][6][6][64] -> feat[:, :1024] (NCHW-flatten), k3 s1, K=576 (HWIO)
DEV v8h ld8_c3(const _Float16* a2, int r, int kk) {
  int n = r >> 4, s2 = r & 15;
  int oy = s2 >> 2, ox = s2 & 3;
  int kcell = kk >> 6, c0 = kk & 63;
  int ky = kcell / 3, kx = kcell % 3;
  return *(const v8h*)(a2 + (((size_t)n * 6 + oy + ky) * 6 + (ox + kx)) * 64 + c0);
}

__global__ __launch_bounds__(128)
void k_conv3(const _Float16* __restrict__ a2, const _Float16* __restrict__ W,
             const float* __restrict__ bias, _Float16* __restrict__ feat) {
  int lane = threadIdx.x & 31;
  int mt = blockIdx.y * 4 + (threadIdx.x >> 5);  // covers images 2*mt, 2*mt+1
  int row0 = mt * 32, n0 = blockIdx.x * 32;
  int ncol = lane & 15, m0 = (lane & 16) ? 8 : 0;
  int r0 = row0 + (lane & 15), r1 = r0 + 16;
  float bv0 = bias[n0 + ncol], bv1 = bias[n0 + 16 + ncol];
  v8f c00, c01, c10, c11;
#pragma unroll
  for (int j = 0; j < 8; j++) { c00[j] = bv0; c01[j] = bv1; c10[j] = bv0; c11[j] = bv1; }
  for (int kb = 0; kb < 576; kb += 32) {
    int k1 = kb + ((lane & 16) ? 8 : 0);
    v16h a0 = mk16(ld8_c3(a2, r0, k1), ld8_c3(a2, r0, k1 + 16));
    v16h a1 = mk16(ld8_c3(a2, r1, k1), ld8_c3(a2, r1, k1 + 16));
    v16h b0 = frag_rm(W, 576, n0,      kb, lane);
    v16h b1 = frag_rm(W, 576, n0 + 16, kb, lane);
    c00 = wmma16(a0, b0, c00);
    c01 = wmma16(a0, b1, c01);
    c10 = wmma16(a1, b0, c10);
    c11 = wmma16(a1, b1, c11);
  }
  // feat[n][ch*16 + spatial]: 8 contiguous halves per lane -> b128 stores
  v8f cs[4] = {c00, c01, c10, c11};
#pragma unroll
  for (int q = 0; q < 4; q++) {
    int sm = q >> 1, sn = q & 1;
    int img = 2 * mt + sm;
    int ch = n0 + sn * 16 + ncol;
    v8h o;
#pragma unroll
    for (int j = 0; j < 8; j++) {
      float v = cs[q][j] > 0.f ? cs[q][j] : 0.f;
      o[j] = (_Float16)v;
    }
    *(v8h*)(feat + (size_t)img * FEATD + ch * 16 + m0) = o;
  }
}

// ---------------- sequential 2-layer x 2-stack LSTM (one workgroup) ----------------
__global__ __launch_bounds__(256)
void k_lstm(const float* __restrict__ G0, const float* __restrict__ G1, // [1024,GPAD]
            const _Float16* __restrict__ whh0s, const _Float16* __restrict__ wih1s,
            const _Float16* __restrict__ whh1s,
            const _Float16* __restrict__ whh0r, const _Float16* __restrict__ wih1r,
            const _Float16* __restrict__ whh1r,
            const float* __restrict__ b1s, const float* __restrict__ b1r,
            float* __restrict__ cst,      // [2][32][100] cell states
            float* __restrict__ lastOut)  // [2][32][100] last h of layer1 per stack
{
  __shared__ _Float16 gb[32 * G4];
  __shared__ _Float16 h0s[32 * HPAD];
  __shared__ _Float16 h1s[32 * HPAD];
  int tid = threadIdx.x, lane = tid & 31, wv = tid >> 5;

  for (int i = tid; i < 32 * HPAD; i += 256) { h0s[i] = (_Float16)0.f; h1s[i] = (_Float16)0.f; }
  for (int i = tid; i < 2 * 32 * HH; i += 256) cst[i] = 0.f;
  __syncthreads();

  for (int s = 0; s < 2; s++) {
    const float*    G    = s ? G1 : G0;
    const _Float16* whh0 = s ? whh0r : whh0s;
    const _Float16* wih1 = s ? wih1r : wih1s;
    const _Float16* whh1 = s ? whh1r : whh1s;
    const float*    b1   = s ? b1r   : b1s;
    float* c0 = cst;
    float* c1 = cst + 32 * HH;

    for (int t = 0; t < 32; t++) {
      // --- layer0 gates: g = G[:,t,:] + h0 @ whh0^T ---
      for (int tile = wv; tile < 50; tile += 8) {
        int m0t = (tile & 1) * 16, n0 = (tile >> 1) * 16;
        int ncol = lane & 15, mo = (lane & 16) ? 8 : 0;
        v8f c;
#pragma unroll
        for (int j = 0; j < 8; j++)
          c[j] = G[((size_t)((m0t + mo + j) * 32 + t)) * GPAD + n0 + ncol];
#pragma unroll
        for (int kb = 0; kb < HPAD; kb += 32) {
          v16h a = frag_rm(h0s, HPAD, m0t, kb, lane);
          v16h b = frag_rm(whh0, HPAD, n0, kb, lane);
          c = wmma16(a, b, c);
        }
#pragma unroll
        for (int j = 0; j < 8; j++)
          gb[(m0t + mo + j) * G4 + n0 + ncol] = (_Float16)c[j];
      }
      __syncthreads();
      // --- layer0 elementwise (torch gate order i,f,g,o) ---
      for (int i = tid; i < 32 * HH; i += 256) {
        int b = i / HH, hh = i % HH;
        float gi = sigf((float)gb[b * G4 + hh]);
        float gf = sigf((float)gb[b * G4 + HH + hh]);
        float gg = tanh_fast((float)gb[b * G4 + 2 * HH + hh]);
        float go = sigf((float)gb[b * G4 + 3 * HH + hh]);
        float cc = gf * c0[i] + gi * gg;
        c0[i] = cc;
        h0s[b * HPAD + hh] = (_Float16)(go * tanh_fast(cc));
      }
      __syncthreads();
      // --- layer1 gates: g = bias1 + h0 @ wih1^T + h1 @ whh1^T ---
      for (int tile = wv; tile < 50; tile += 8) {
        int m0t = (tile & 1) * 16, n0 = (tile >> 1) * 16;
        int ncol = lane & 15, mo = (lane & 16) ? 8 : 0;
        float bv = b1[n0 + ncol];
        v8f c;
#pragma unroll
        for (int j = 0; j < 8; j++) c[j] = bv;
#pragma unroll
        for (int kb = 0; kb < HPAD; kb += 32) {
          v16h a = frag_rm(h0s, HPAD, m0t, kb, lane);
          v16h b = frag_rm(wih1, HPAD, n0, kb, lane);
          c = wmma16(a, b, c);
        }
#pragma unroll
        for (int kb = 0; kb < HPAD; kb += 32) {
          v16h a = frag_rm(h1s, HPAD, m0t, kb, lane);
          v16h b = frag_rm(whh1, HPAD, n0, kb, lane);
          c = wmma16(a, b, c);
        }
#pragma unroll
        for (int j = 0; j < 8; j++)
          gb[(m0t + mo + j) * G4 + n0 + ncol] = (_Float16)c[j];
      }
      __syncthreads();
      // --- layer1 elementwise ---
      for (int i = tid; i < 32 * HH; i += 256) {
        int b = i / HH, hh = i % HH;
        float gi = sigf((float)gb[b * G4 + hh]);
        float gf = sigf((float)gb[b * G4 + HH + hh]);
        float gg = tanh_fast((float)gb[b * G4 + 2 * HH + hh]);
        float go = sigf((float)gb[b * G4 + 3 * HH + hh]);
        float cc = gf * c1[i] + gi * gg;
        c1[i] = cc;
        h1s[b * HPAD + hh] = (_Float16)(go * tanh_fast(cc));
      }
      __syncthreads();
    }
    // record layer-1 last hidden; hn persists in LDS, cn in global cst, into stack 1
    for (int i = tid; i < 32 * HH; i += 256) {
      int b = i / HH, hh = i % HH;
      lastOut[s * 32 * HH + i] = (float)h1s[b * HPAD + hh];
    }
    __syncthreads();
  }
}

// ---------------- head: concat(out, out_s) -> fc1 relu -> fc2 ----------------
__global__ __launch_bounds__(256)
void k_head(const float* __restrict__ lastOut, // [2][32][100] (0=rnns/out_s, 1=rnn/out)
            const _Float16* __restrict__ fc1w, const float* __restrict__ fc1b,
            const _Float16* __restrict__ fc2w, const float* __restrict__ fc2b,
            float* __restrict__ out) {
  __shared__ _Float16 xin[32 * FC1K];
  __shared__ _Float16 hid[32 * 512];
  int tid = threadIdx.x, lane = tid & 31, wv = tid >> 5;
  for (int i = tid; i < 32 * FC1K; i += 256) {
    int b = i / FC1K, j = i % FC1K;
    float v = 0.f;
    if (j < 100)      v = lastOut[32 * HH + b * HH + j];   // out (rnn stack)
    else if (j < 200) v = lastOut[b * HH + (j - 100)];     // out_s (rnns stack)
    xin[i] = (_Float16)v;
  }
  __syncthreads();
  // fc1: [32,224] x [512,224]^T, relu
  for (int tile = wv; tile < 64; tile += 8) {
    int m0t = (tile & 1) * 16, n0 = (tile >> 1) * 16;
    int ncol = lane & 15, mo = (lane & 16) ? 8 : 0;
    float bv = fc1b[n0 + ncol];
    v8f c;
#pragma unroll
    for (int j = 0; j < 8; j++) c[j] = bv;
#pragma unroll
    for (int kb = 0; kb < FC1K; kb += 32) {
      v16h a = frag_rm(xin, FC1K, m0t, kb, lane);
      v16h b = frag_rm(fc1w, FC1K, n0, kb, lane);
      c = wmma16(a, b, c);
    }
#pragma unroll
    for (int j = 0; j < 8; j++) {
      float v = c[j] > 0.f ? c[j] : 0.f;
      hid[(m0t + mo + j) * 512 + n0 + ncol] = (_Float16)v;
    }
  }
  __syncthreads();
  // fc2: [32,512] x [144,512]^T, store cols < 130
  for (int tile = wv; tile < 2 * (OUTPAD / 16); tile += 8) {
    int m0t = (tile & 1) * 16, n0 = (tile >> 1) * 16;
    int ncol = lane & 15, mo = (lane & 16) ? 8 : 0;
    int nn = n0 + ncol;
    float bv = (nn < OUTN) ? fc2b[nn] : 0.f;
    v8f c;
#pragma unroll
    for (int j = 0; j < 8; j++) c[j] = bv;
#pragma unroll
    for (int kb = 0; kb < 512; kb += 32) {
      v16h a = frag_rm(hid, 512, m0t, kb, lane);
      v16h b = frag_rm(fc2w, 512, n0, kb, lane);
      c = wmma16(a, b, c);
    }
    if (nn < OUTN) {
#pragma unroll
      for (int j = 0; j < 8; j++)
        out[(size_t)(m0t + mo + j) * OUTN + nn] = c[j];
    }
  }
}

// ---------------- host ----------------
extern "C" void kernel_launch(void* const* d_in, const int* in_sizes, int n_in,
                              void* d_out, int out_size, void* d_ws, size_t ws_size,
                              hipStream_t stream) {
  (void)in_sizes; (void)n_in; (void)out_size; (void)ws_size;
  const float* obs    = (const float*)d_in[0];
  const float* data   = (const float*)d_in[1];
  const float* c1wf   = (const float*)d_in[2];
  const float* c1b    = (const float*)d_in[3];
  const float* c2wf   = (const float*)d_in[4];
  const float* c2b    = (const float*)d_in[5];
  const float* c3wf   = (const float*)d_in[6];
  const float* c3b    = (const float*)d_in[7];
  const float* l1wf   = (const float*)d_in[8];
  const float* l1b    = (const float*)d_in[9];
  const float* l2wf   = (const float*)d_in[10];
  const float* l2b    = (const float*)d_in[11];
  const float* s_wih0 = (const float*)d_in[12];
  const float* s_whh0 = (const float*)d_in[13];
  const float* s_bih0 = (const float*)d_in[14];
  const float* s_bhh0 = (const float*)d_in[15];
  const float* s_wih1 = (const float*)d_in[16];
  const float* s_whh1 = (const float*)d_in[17];
  const float* s_bih1 = (const float*)d_in[18];
  const float* s_bhh1 = (const float*)d_in[19];
  const float* r_wih0 = (const float*)d_in[20];
  const float* r_whh0 = (const float*)d_in[21];
  const float* r_bih0 = (const float*)d_in[22];
  const float* r_bhh0 = (const float*)d_in[23];
  const float* r_wih1 = (const float*)d_in[24];
  const float* r_whh1 = (const float*)d_in[25];
  const float* r_bih1 = (const float*)d_in[26];
  const float* r_bhh1 = (const float*)d_in[27];
  const float* fc1wf  = (const float*)d_in[28];
  const float* fc1b   = (const float*)d_in[29];
  const float* fc2wf  = (const float*)d_in[30];
  const float* fc2b   = (const float*)d_in[31];
  float* out = (float*)d_out;

  // ---- workspace carve ----
  size_t off = 0;
  auto awh = [&](size_t elems) {
    void* p = (char*)d_ws + off;
    off += ((elems * sizeof(_Float16)) + 255) & ~(size_t)255;
    return (_Float16*)p;
  };
  auto awf = [&](size_t elems) {
    void* p = (char*)d_ws + off;
    off += ((elems * sizeof(float)) + 255) & ~(size_t)255;
    return (float*)p;
  };
  _Float16* data16 = awh((size_t)NTOT * DPAD);
  _Float16* l1w16  = awh((size_t)1024 * DPAD);
  _Float16* l2w16  = awh((size_t)512 * 1024);
  _Float16* c1w16  = awh((size_t)32 * 192);
  _Float16* c2w16  = awh((size_t)64 * 512);
  _Float16* c3w16  = awh((size_t)64 * 576);
  _Float16* wihs0  = awh((size_t)GPAD * FEATD);
  _Float16* wihr0  = awh((size_t)GPAD * FEATD);
  _Float16* whhs0  = awh((size_t)G4 * HPAD);
  _Float16* wihs1  = awh((size_t)G4 * HPAD);
  _Float16* whhs1  = awh((size_t)G4 * HPAD);
  _Float16* whhr0  = awh((size_t)G4 * HPAD);
  _Float16* wihr1  = awh((size_t)G4 * HPAD);
  _Float16* whhr1  = awh((size_t)G4 * HPAD);
  _Float16* fc1w16 = awh((size_t)512 * FC1K);
  _Float16* fc2w16 = awh((size_t)OUTPAD * 512);
  _Float16* a1     = awh((size_t)NTOT * 15 * 15 * 32);
  _Float16* a2     = awh((size_t)NTOT * 6 * 6 * 64);
  _Float16* feat   = awh((size_t)NTOT * FEATD);
  _Float16* d1     = awh((size_t)NTOT * 1024);
  float* G0      = awf((size_t)1024 * GPAD);
  float* G1      = awf((size_t)1024 * GPAD);
  float* bias_s0 = awf(GPAD);
  float* bias_s1 = awf(G4);
  float* bias_r0 = awf(GPAD);
  float* bias_r1 = awf(G4);
  float* cst     = awf(2 * 32 * HH);
  float* lastOut = awf(2 * 32 * HH);

  // ---- pack / convert ----
  k_cvt_pad<<<2048, 256, 0, stream>>>(data16, data, NTOT, NTOT, DPAD, DDIM);
  k_cvt_pad<<<1024, 256, 0, stream>>>(l1w16, l1wf, 1024, 1024, DPAD, DDIM);
  k_cvt_pad<<<512, 256, 0, stream>>>(l2w16, l2wf, 512, 512, 1024, 1024);
  k_cvt_pad<<<32, 256, 0, stream>>>(c1w16, c1wf, 32, 32, 192, 192);
  k_pack_hwio<<<128, 256, 0, stream>>>(c2w16, c2wf, 64, 32, 4, 4);
  k_pack_hwio<<<160, 256, 0, stream>>>(c3w16, c3wf, 64, 64, 3, 3);
  k_cvt_pad<<<1024, 256, 0, stream>>>(wihs0, s_wih0, GPAD, G4, FEATD, FEATD);
  k_cvt_pad<<<1024, 256, 0, stream>>>(wihr0, r_wih0, GPAD, G4, FEATD, FEATD);
  k_cvt_pad<<<200, 256, 0, stream>>>(whhs0, s_whh0, G4, G4, HPAD, HH);
  k_cvt_pad<<<200, 256, 0, stream>>>(wihs1, s_wih1, G4, G4, HPAD, HH);
  k_cvt_pad<<<200, 256, 0, stream>>>(whhs1, s_whh1, G4, G4, HPAD, HH);
  k_cvt_pad<<<200, 256, 0, stream>>>(whhr0, r_whh0, G4, G4, HPAD, HH);
  k_cvt_pad<<<200, 256, 0, stream>>>(wihr1, r_wih1, G4, G4, HPAD, HH);
  k_cvt_pad<<<200, 256, 0, stream>>>(whhr1, r_whh1, G4, G4, HPAD, HH);
  k_cvt_pad<<<448, 256, 0, stream>>>(fc1w16, fc1wf, 512, 512, FC1K, 200);
  k_cvt_pad<<<288, 256, 0, stream>>>(fc2w16, fc2wf, OUTPAD, OUTN, 512, 512);
  k_bias_pad<<<2, 256, 0, stream>>>(bias_s0, s_bih0, s_bhh0, G4, GPAD);
  k_bias_pad<<<2, 256, 0, stream>>>(bias_s1, s_bih1, s_bhh1, G4, G4);
  k_bias_pad<<<2, 256, 0, stream>>>(bias_r0, r_bih0, r_bhh0, G4, GPAD);
  k_bias_pad<<<2, 256, 0, stream>>>(bias_r1, r_bih1, r_bhh1, G4, G4);

  // ---- conv stack (implicit im2col WMMA, 2x2 blocked) ----
  // conv1: M=460800 (14400 32-row blocks), N=32 (one 32-col block), K=192
  k_conv1<<<dim3(1, 3600), 128, 0, stream>>>(obs, c1w16, c1b, a1);
  // conv2: M=73728 (2304 blocks), N=64 (2), K=512
  k_conv2<<<dim3(2, 576), 128, 0, stream>>>(a1, c2w16, c2b, a2);
  // conv3: M=32768 (1024 blocks), N=64 (2), K=576 -> feat[:, :1024]
  k_conv3<<<dim3(2, 256), 128, 0, stream>>>(a2, c3w16, c3b, feat);

  // ---- MLP branch ----
  // lin1: [2048,1504] x [1024,1504]^T relu -> d1
  k_gemm2<<<dim3(32, 16), 128, 0, stream>>>(data16, DPAD, l1w16, DPAD, l1b,
                                            d1, 1024, 64, DPAD, 1);
  // lin2: [2048,1024] x [512,1024]^T -> feat[:, 1024:]
  k_gemm2<<<dim3(16, 16), 128, 0, stream>>>(d1, 1024, l2w16, 1024, l2b,
                                            feat + 1024, FEATD, 64, 1024, 0);

  // ---- LSTM input projections (x @ wih0^T + bih0 + bhh0), N padded to 416 ----
  k_gemm2<<<dim3(13, 8), 128, 0, stream>>>(feat, FEATD, wihs0, FEATD, bias_s0,
                                           G0, GPAD, 32, FEATD, 2);
  k_gemm2<<<dim3(13, 8), 128, 0, stream>>>(feat + (size_t)1024 * FEATD, FEATD,
                                           wihr0, FEATD, bias_r0,
                                           G1, GPAD, 32, FEATD, 2);

  // ---- sequential recurrence (both stacks, h carry in LDS, c in global cst) ----
  k_lstm<<<1, 256, 0, stream>>>(G0, G1, whhs0, wihs1, whhs1, whhr0, wihr1, whhr1,
                                bias_s1, bias_r1, cst, lastOut);

  // ---- head ----
  k_head<<<1, 256, 0, stream>>>(lastOut, fc1w16, fc1b, fc2w16, fc2b, out);
}
